// ASAP_26036091748783
// MI455X (gfx1250) — matrix operand order repaired and checked by reading, not compiled, not run on hardware
//
#include <hip/hip_runtime.h>
#include <cstddef>
#include <cstdint>

#define DEV __device__ __forceinline__

constexpr int   N_    = 4096;
constexpr int   B_    = 16;
constexpr int   NG_   = 256;
constexpr int   E_    = 131072;
constexpr int   EPL_  = E_ + N_;        // real edges + self loops
constexpr int   HID_  = 256;
constexpr int   KPER_ = 205;            // ceil(0.8*256)
constexpr int   K_    = B_ * KPER_;     // 3280 = 205*16
constexpr int   NCLS_ = 128;
constexpr int   KCH_  = (K_ + 255) / 256; // 13
constexpr float NEG_  = 0.2f;

typedef __attribute__((ext_vector_type(16))) __bf16 v16bf;
typedef __attribute__((ext_vector_type(8)))  float  v8f;
union v16bf_u { v16bf v; __bf16 e[16]; };
union v8f_u   { v8f   v; float  e[8];  };

// native f32->bf16 (fptrunc, RNE) -- lets the backend pick v_cvt_pk_bf16_f32-class ops
DEV __bf16 f2bf(float f) { return (__bf16)f; }

// order-preserving float->uint key (for atomicMax-based segment max)
DEV unsigned fkey(float f)  { unsigned u = __float_as_uint(f); return (u & 0x80000000u) ? ~u : (u | 0x80000000u); }
DEV float    funkey(unsigned k){ unsigned u = (k & 0x80000000u) ? (k & 0x7FFFFFFFu) : ~k; return __uint_as_float(u); }
DEV int rowOf(int e, const int* src){ return e < E_ ? src[e] : e - E_; }
DEV int colOf(int e, const int* dst){ return e < E_ ? dst[e] : e - E_; }

DEV void cvt8(v16bf_u& o, int base, float4 a, float4 b) {
  o.e[base + 0] = f2bf(a.x); o.e[base + 1] = f2bf(a.y);
  o.e[base + 2] = f2bf(a.z); o.e[base + 3] = f2bf(a.w);
  o.e[base + 4] = f2bf(b.x); o.e[base + 5] = f2bf(b.y);
  o.e[base + 6] = f2bf(b.z); o.e[base + 7] = f2bf(b.w);
}

// ---------------------------------------------------------------- GEMM (bf16 WMMA, f32 accum)
// C[m,n] = act( Aeff1[m,:]@B1[:,n] (+ A2[m,:]@B2[:,n]) + bias[n] ),  Aeff1 = tA1 ? A1^T : A1
// block = 256 thr = 8 waves; wave -> 16x64 output strip (4 WMMA n-tiles, A frag reused);
// grid.x = ceil(M/32). M multiple of 16; Ncols in {128,256} so active waves own a full strip.
__global__ void __launch_bounds__(256) k_gemm(
    const float* __restrict__ A1, int lda1, int tA1,
    const float* __restrict__ B1, int ldb1,
    const float* __restrict__ A2, int lda2,
    const float* __restrict__ B2, int ldb2,
    const float* __restrict__ bias,
    float* __restrict__ C, int ldc,
    int M, int Kc, int Ncols, int do_relu)
{
  const int lane  = threadIdx.x & 31;
  const int wave  = threadIdx.x >> 5;
  const int g     = lane >> 4;          // lane half
  const int l16   = lane & 15;
  const int mtile = blockIdx.x * 2 + (wave >> 2);
  const int nbase = (wave & 3) * 64;
  if (mtile * 16 >= M) return;          // wave-uniform
  if (nbase >= Ncols)  return;          // wave-uniform

  v8f_u acc[4];
  #pragma unroll
  for (int t = 0; t < 4; ++t)
    #pragma unroll
    for (int r = 0; r < 8; ++r) acc[t].e[r] = 0.f;

  const int m = mtile * 16 + l16;       // A row owned by this lane
  const int kmain = Kc & ~31;           // unguarded main loop extent

  // ---------------- main loop: no guards, vector loads on the A side
  for (int k0 = 0; k0 < kmain; k0 += 32) {
    v16bf_u a1;
    if (tA1) {
      #pragma unroll
      for (int i = 0; i < 16; ++i) {
        int k = k0 + ((i < 8) ? (g * 8 + i) : (16 + g * 8 + (i - 8)));
        a1.e[i] = f2bf(A1[(size_t)k * lda1 + m]);
      }
    } else {
      const float* ap = A1 + (size_t)m * lda1 + k0 + g * 8;
      cvt8(a1, 0, *(const float4*)(ap),      *(const float4*)(ap + 4));
      cvt8(a1, 8, *(const float4*)(ap + 16), *(const float4*)(ap + 20));
    }
    v16bf_u a2f;
    if (A2) {
      const float* ap = A2 + (size_t)m * lda2 + k0 + g * 8;
      cvt8(a2f, 0, *(const float4*)(ap),      *(const float4*)(ap + 4));
      cvt8(a2f, 8, *(const float4*)(ap + 16), *(const float4*)(ap + 20));
    }
    #pragma unroll
    for (int t = 0; t < 4; ++t) {
      int n = nbase + t * 16 + l16;
      v16bf_u b1;
      #pragma unroll
      for (int i = 0; i < 16; ++i)
        b1.e[i] = f2bf(B1[(size_t)(k0 + ((i < 8) ? (g * 8 + i) : (16 + g * 8 + (i - 8)))) * ldb1 + n]);
      acc[t].v = __builtin_amdgcn_wmma_f32_16x16x32_bf16(
          false, a1.v, false, b1.v, (short)0, acc[t].v, false, false);
      if (A2) {
        v16bf_u b2;
        #pragma unroll
        for (int i = 0; i < 16; ++i)
          b2.e[i] = f2bf(B2[(size_t)(k0 + ((i < 8) ? (g * 8 + i) : (16 + g * 8 + (i - 8)))) * ldb2 + n]);
        acc[t].v = __builtin_amdgcn_wmma_f32_16x16x32_bf16(
            false, a2f.v, false, b2.v, (short)0, acc[t].v, false, false);
      }
    }
  }

  // ---------------- branchless tail (at most one step): clamped address + value select
  if (kmain < Kc) {
    const int k0 = kmain;
    v16bf_u a1;
    #pragma unroll
    for (int i = 0; i < 16; ++i) {
      int k  = k0 + ((i < 8) ? (g * 8 + i) : (16 + g * 8 + (i - 8)));
      int kk = k < Kc ? k : Kc - 1;
      float v = tA1 ? A1[(size_t)kk * lda1 + m] : A1[(size_t)m * lda1 + kk];
      a1.e[i] = f2bf(k < Kc ? v : 0.f);
    }
    v16bf_u a2f;
    if (A2) {
      #pragma unroll
      for (int i = 0; i < 16; ++i) {
        int k  = k0 + ((i < 8) ? (g * 8 + i) : (16 + g * 8 + (i - 8)));
        int kk = k < Kc ? k : Kc - 1;
        float v = A2[(size_t)m * lda2 + kk];
        a2f.e[i] = f2bf(k < Kc ? v : 0.f);
      }
    }
    #pragma unroll
    for (int t = 0; t < 4; ++t) {
      int n = nbase + t * 16 + l16;
      v16bf_u b1;
      #pragma unroll
      for (int i = 0; i < 16; ++i) {
        int k  = k0 + ((i < 8) ? (g * 8 + i) : (16 + g * 8 + (i - 8)));
        int kk = k < Kc ? k : Kc - 1;
        float v = B1[(size_t)kk * ldb1 + n];
        b1.e[i] = f2bf(k < Kc ? v : 0.f);
      }
      acc[t].v = __builtin_amdgcn_wmma_f32_16x16x32_bf16(
          false, a1.v, false, b1.v, (short)0, acc[t].v, false, false);
      if (A2) {
        v16bf_u b2;
        #pragma unroll
        for (int i = 0; i < 16; ++i) {
          int k  = k0 + ((i < 8) ? (g * 8 + i) : (16 + g * 8 + (i - 8)));
          int kk = k < Kc ? k : Kc - 1;
          float v = B2[(size_t)kk * ldb2 + n];
          b2.e[i] = f2bf(k < Kc ? v : 0.f);
        }
        acc[t].v = __builtin_amdgcn_wmma_f32_16x16x32_bf16(
            false, a2f.v, false, b2.v, (short)0, acc[t].v, false, false);
      }
    }
  }

  // C layout: v8f elem r, lane (g,l16) -> row = mtile*16 + g*8 + r, col = n
  #pragma unroll
  for (int t = 0; t < 4; ++t) {
    int n = nbase + t * 16 + l16;
    float bv = bias ? bias[n] : 0.f;
    #pragma unroll
    for (int r = 0; r < 8; ++r) {
      int row = mtile * 16 + g * 8 + r;
      float v = acc[t].e[r] + bv;
      if (do_relu) v = v > 0.f ? v : 0.f;
      C[(size_t)row * ldc + n] = v;
    }
  }
}

// ---------------------------------------------------------------- elementwise / graph kernels
__global__ void k_embed(const int* __restrict__ xt, const int* __restrict__ nd,
                        const float* __restrict__ et, const float* __restrict__ ed,
                        float* __restrict__ x) {
  int n = blockIdx.x, c = threadIdx.x;
  x[(size_t)n * HID_ + c] = et[(size_t)xt[n] * HID_ + c] + ed[(size_t)nd[n] * HID_ + c];
}

__global__ void k_edge_agg(const int* __restrict__ src, const int* __restrict__ dst,
                           const float* __restrict__ x, float* __restrict__ agg,
                           float* __restrict__ cnt) {
  int e = blockIdx.x, t = threadIdx.x;
  int s = src[e], d = dst[e];
  atomicAdd(&agg[(size_t)d * HID_ + t], x[(size_t)s * HID_ + t]);
  if (t == 0) atomicAdd(&cnt[d], 1.0f);
}

__global__ void k_mean(float* __restrict__ agg, const float* __restrict__ cnt, int total) {
  int i = blockIdx.x * 256 + threadIdx.x;
  if (i < total) agg[i] /= fmaxf(cnt[i >> 8], 1.0f);   // HID_=256
}

__global__ void k_mx(const int* __restrict__ src, const int* __restrict__ dst,
                     const float* __restrict__ x, unsigned* __restrict__ mxu) {
  int e = blockIdx.x, t = threadIdx.x;
  int r = rowOf(e, src), c = colOf(e, dst);
  atomicMax(&mxu[(size_t)c * HID_ + t], fkey(x[(size_t)r * HID_ + t]));
}

__global__ void k_unkey(const unsigned* __restrict__ mxu, float* __restrict__ mxf, int total) {
  int i = blockIdx.x * 256 + threadIdx.x;
  if (i < total) mxf[i] = funkey(mxu[i]);
}

__global__ void k_dot1(const float* __restrict__ x, const float* __restrict__ w,
                       float* __restrict__ out, const float* __restrict__ addp) {
  __shared__ float sd[256];
  int n = blockIdx.x, t = threadIdx.x;
  sd[t] = x[(size_t)n * HID_ + t] * w[t];
  __syncthreads();
  for (int s = 128; s > 0; s >>= 1) { if (t < s) sd[t] += sd[t + s]; __syncthreads(); }
  if (t == 0) out[n] = sd[0] + (addp ? addp[0] : 0.f);
}

__global__ void k_smax(const int* __restrict__ src, const int* __restrict__ dst,
                       const float* __restrict__ an, const float* __restrict__ cn,
                       float* __restrict__ sE, unsigned* __restrict__ smaxu) {
  int e = blockIdx.x * 256 + threadIdx.x;
  if (e >= EPL_) return;
  int r = rowOf(e, src), c = colOf(e, dst);
  float s = an[c] + cn[r];
  s = s > 0.f ? s : NEG_ * s;
  sE[e] = s;
  atomicMax(&smaxu[c], fkey(s));
}

__global__ void k_exp(const int* __restrict__ src, const int* __restrict__ dst,
                      const float* __restrict__ sE, const unsigned* __restrict__ smaxu,
                      float* __restrict__ eE, float* __restrict__ denom) {
  int e = blockIdx.x * 256 + threadIdx.x;
  if (e >= EPL_) return;
  int c = colOf(e, dst);
  (void)src;
  float ee = expf(sE[e] - funkey(smaxu[c]));
  eE[e] = ee;
  atomicAdd(&denom[c], ee);
}

__global__ void k_p(const int* __restrict__ dst, const float* __restrict__ eE,
                    const float* __restrict__ denom, float* __restrict__ pE) {
  int e = blockIdx.x * 256 + threadIdx.x;
  if (e >= EPL_) return;
  pE[e] = eE[e] / denom[colOf(e, dst)];
}

__global__ void k_xnew(const int* __restrict__ src, const int* __restrict__ dst,
                       const float* __restrict__ x, const float* __restrict__ pE,
                       float* __restrict__ xnew) {
  int e = blockIdx.x, t = threadIdx.x;
  int r = rowOf(e, src), c = colOf(e, dst);
  atomicAdd(&xnew[(size_t)c * HID_ + t], x[(size_t)r * HID_ + t] * pE[e]);
}

__global__ void k_fitacc(const int* __restrict__ src, const int* __restrict__ dst,
                         const float* __restrict__ av, const float* __restrict__ bm,
                         float* __restrict__ fitacc) {
  int e = blockIdx.x * 256 + threadIdx.x;
  if (e >= EPL_) return;
  int r = rowOf(e, src), c = colOf(e, dst);
  atomicAdd(&fitacc[c], av[r] - bm[c]);
}

__global__ void k_fitness(const float* __restrict__ fitacc, const float* __restrict__ w1d,
                          const float* __restrict__ b1, float* __restrict__ fitn) {
  int n = blockIdx.x * 256 + threadIdx.x;
  if (n >= N_) return;
  float f = fitacc[n] + w1d[n] + b1[0];
  fitn[n] = 1.f / (1.f + expf(-f));
}

// exact top-k as a SET (downstream is order-invariant within a graph): rank by (value desc, index asc)
__global__ void k_topk(const float* __restrict__ fitn, int* __restrict__ selcnt,
                       int* __restrict__ perm, int* __restrict__ pos,
                       float* __restrict__ fitsel) {
  __shared__ float f[256];
  int b = blockIdx.x, i = threadIdx.x;
  int n = b * NG_ + i;
  f[i] = fitn[n];
  __syncthreads();
  float fi = f[i];
  int rank = 0;
  for (int j = 0; j < NG_; ++j)
    rank += (f[j] > fi) || (f[j] == fi && j < i);
  if (rank < KPER_) {
    int k = b * KPER_ + atomicAdd(&selcnt[b], 1);
    perm[k] = n; pos[n] = k; fitsel[k] = fi;
  } else {
    pos[n] = -1;
  }
}

__global__ void k_xsel(const float* __restrict__ xnew, const int* __restrict__ perm,
                       const float* __restrict__ fitsel, float* __restrict__ xsel) {
  int k = blockIdx.x, t = threadIdx.x;
  xsel[(size_t)k * HID_ + t] = xnew[(size_t)perm[k] * HID_ + t] * fitsel[k];
}

// S_sel[r, pos[c]] += p_e  (dense N x K)
__global__ void k_Sscatter(const int* __restrict__ src, const int* __restrict__ dst,
                           const int* __restrict__ pos, const float* __restrict__ pE,
                           float* __restrict__ Ssel) {
  int e = blockIdx.x * 256 + threadIdx.x;
  if (e >= EPL_) return;
  int r = rowOf(e, src), c = colOf(e, dst);
  int kp = pos[c];
  if (kp >= 0) atomicAdd(&Ssel[(size_t)r * K_ + kp], pE[e]);
}

// CSR construction helpers (arr = src for row-CSR, dst for col-CSR)
__global__ void k_hist(const int* __restrict__ arr, int* __restrict__ cnt) {
  int e = blockIdx.x * 256 + threadIdx.x;
  if (e >= EPL_) return;
  int n = e < E_ ? arr[e] : e - E_;
  atomicAdd(&cnt[n], 1);
}
__global__ void k_scan(const int* __restrict__ cnt, int* __restrict__ ptr,
                       int* __restrict__ cur, int n) {
  __shared__ int part[256];
  int t = threadIdx.x;
  int chunk = (n + 255) / 256;
  int s = 0;
  for (int i = 0; i < chunk; ++i) { int idx = t * chunk + i; if (idx < n) s += cnt[idx]; }
  part[t] = s;
  __syncthreads();
  if (t == 0) {
    int acc = 0;
    for (int i = 0; i < 256; ++i) { int v = part[i]; part[i] = acc; acc += v; }
    ptr[n] = acc;
  }
  __syncthreads();
  int acc = part[t];
  for (int i = 0; i < chunk; ++i) {
    int idx = t * chunk + i;
    if (idx < n) { ptr[idx] = acc; cur[idx] = acc; acc += cnt[idx]; }
  }
}
__global__ void k_scat(const int* __restrict__ arr, int* __restrict__ cur,
                       int* __restrict__ edges) {
  int e = blockIdx.x * 256 + threadIdx.x;
  if (e >= EPL_) return;
  int n = e < E_ ? arr[e] : e - E_;
  int slot = atomicAdd(&cur[n], 1);
  edges[slot] = e;
}

// M[i,:] = sum over edges with row==i of S_sel[col_e, :]
__global__ void __launch_bounds__(256) k_AM(const int* __restrict__ rptr,
                                            const int* __restrict__ redges,
                                            const int* __restrict__ dst,
                                            const float* __restrict__ Ssel,
                                            float* __restrict__ Mb) {
  int i = blockIdx.x, t = threadIdx.x;
  float acc[KCH_];
  #pragma unroll
  for (int j = 0; j < KCH_; ++j) acc[j] = 0.f;
  for (int idx = rptr[i]; idx < rptr[i + 1]; ++idx) {
    int e = redges[idx];
    int c = colOf(e, dst);
    const float* srow = &Ssel[(size_t)c * K_];
    #pragma unroll
    for (int j = 0; j < KCH_; ++j) { int col = t + j * 256; if (col < K_) acc[j] += srow[col]; }
  }
  #pragma unroll
  for (int j = 0; j < KCH_; ++j) { int col = t + j * 256; if (col < K_) Mb[(size_t)i * K_ + col] = acc[j]; }
}

// A2[k1,:] = sum over edges into c=perm[k1] of p_e * M[row_e, :]; diag forced to 1
__global__ void __launch_bounds__(256) k_A2(const int* __restrict__ perm,
                                            const int* __restrict__ cptr,
                                            const int* __restrict__ cedges,
                                            const int* __restrict__ src,
                                            const float* __restrict__ pE,
                                            const float* __restrict__ Mb,
                                            float* __restrict__ A2) {
  int k1 = blockIdx.x, t = threadIdx.x;
  int c = perm[k1];
  float acc[KCH_];
  #pragma unroll
  for (int j = 0; j < KCH_; ++j) acc[j] = 0.f;
  for (int idx = cptr[c]; idx < cptr[c + 1]; ++idx) {
    int e = cedges[idx];
    int r = rowOf(e, src);
    float pe = pE[e];
    const float* mrow = &Mb[(size_t)r * K_];
    #pragma unroll
    for (int j = 0; j < KCH_; ++j) { int col = t + j * 256; if (col < K_) acc[j] += pe * mrow[col]; }
  }
  #pragma unroll
  for (int j = 0; j < KCH_; ++j) {
    int col = t + j * 256;
    if (col < K_) A2[(size_t)k1 * K_ + col] = (col == k1) ? 1.0f : acc[j];
  }
}

// cnt2[j] = nnz of column j of A2 (coalesced: thread j scans down rows)
__global__ void k_colcnt(const float* __restrict__ A2, float* __restrict__ cnt2) {
  int j = blockIdx.x * 256 + threadIdx.x;
  if (j >= K_) return;
  float c = 0.f;
  for (int i = 0; i < K_; ++i) c += (A2[(size_t)i * K_ + j] != 0.f) ? 1.f : 0.f;
  cnt2[j] = c;
}

// mean-pool `rows` consecutive rows per graph into h[:, off:off+256]
__global__ void k_pool(const float* __restrict__ x, int rows, float* __restrict__ h, int off) {
  int b = blockIdx.x, c = threadIdx.x;
  float s = 0.f;
  for (int i = 0; i < rows; ++i) s += x[((size_t)b * rows + i) * HID_ + c];
  h[(size_t)b * 1024 + off + c] = s / (float)rows;
}

// ---------------------------------------------------------------- launcher
extern "C" void kernel_launch(void* const* d_in, const int* in_sizes, int n_in,
                              void* d_out, int out_size, void* d_ws, size_t ws_size,
                              hipStream_t stream) {
  (void)in_sizes; (void)n_in; (void)out_size; (void)ws_size;
  const int*   x_type = (const int*)d_in[0];
  const int*   ndep   = (const int*)d_in[1];
  const int*   src    = (const int*)d_in[2];
  const int*   dst    = src + E_;
  const float* et     = (const float*)d_in[3];
  const float* ed     = (const float*)d_in[4];
  const float* c0_Wr = (const float*)d_in[5],  *c0_br = (const float*)d_in[6],  *c0_Wo = (const float*)d_in[7];
  const float* c1_Wr = (const float*)d_in[8],  *c1_br = (const float*)d_in[9],  *c1_Wo = (const float*)d_in[10];
  const float* c2_Wr = (const float*)d_in[11], *c2_br = (const float*)d_in[12], *c2_Wo = (const float*)d_in[13];
  const float* c3_Wr = (const float*)d_in[14], *c3_br = (const float*)d_in[15], *c3_Wo = (const float*)d_in[16];
  const float* p_Wlin = (const float*)d_in[17], *p_blin = (const float*)d_in[18];
  const float* p_Watt = (const float*)d_in[19], *p_batt = (const float*)d_in[20];
  const float* le_W1 = (const float*)d_in[21], *le_b1 = (const float*)d_in[22];
  const float* le_W2 = (const float*)d_in[23], *le_W3 = (const float*)d_in[24], *le_b3 = (const float*)d_in[25];
  const float* lin1_W = (const float*)d_in[26], *lin1_b = (const float*)d_in[27];
  const float* pred_W = (const float*)d_in[28], *pred_b = (const float*)d_in[29];
  float* out = (float*)d_out;

  // ---- workspace carve (aligned 256B)
  char* w = (char*)d_ws;
  size_t off = 0;
  auto alloc = [&](size_t bytes) -> void* {
    off = (off + 255) & ~(size_t)255;
    void* p = w + off; off += bytes; return p;
  };
  const size_t NH = (size_t)N_ * HID_, KH = (size_t)K_ * HID_, NK = (size_t)N_ * K_, KK = (size_t)K_ * K_;
  float*    x0    = (float*)   alloc(NH * 4);
  float*    x1    = (float*)   alloc(NH * 4);
  float*    x2    = (float*)   alloc(NH * 4);
  float*    agg   = (float*)   alloc(NH * 4);
  float*    cnt   = (float*)   alloc(N_ * 4);
  unsigned* mxu   = (unsigned*)alloc(NH * 4);
  float*    mxf   = (float*)   alloc(NH * 4);
  float*    q     = (float*)   alloc(NH * 4);
  float*    an    = (float*)   alloc(N_ * 4);
  float*    cn    = (float*)   alloc(N_ * 4);
  float*    sE    = (float*)   alloc((size_t)EPL_ * 4);
  float*    eE    = (float*)   alloc((size_t)EPL_ * 4);
  float*    pE    = (float*)   alloc((size_t)EPL_ * 4);
  unsigned* smaxu = (unsigned*)alloc(N_ * 4);
  float*    denom = (float*)   alloc(N_ * 4);
  float*    xnew  = (float*)   alloc(NH * 4);
  float*    av    = (float*)   alloc(N_ * 4);
  float*    bm    = (float*)   alloc(N_ * 4);
  float*    w1d   = (float*)   alloc(N_ * 4);
  float*    fita  = (float*)   alloc(N_ * 4);
  float*    fitn  = (float*)   alloc(N_ * 4);
  int*      selc  = (int*)     alloc(B_ * 4);
  int*      perm  = (int*)     alloc(K_ * 4);
  int*      pos   = (int*)     alloc(N_ * 4);
  float*    fits  = (float*)   alloc(K_ * 4);
  float*    xsel  = (float*)   alloc(KH * 4);
  float*    Ssel  = (float*)   alloc(NK * 4);
  float*    Mb    = (float*)   alloc(NK * 4);
  float*    A2    = (float*)   alloc(KK * 4);
  int*      rcnt  = (int*)     alloc(N_ * 4);
  int*      rptr  = (int*)     alloc((N_ + 1) * 4);
  int*      rcur  = (int*)     alloc(N_ * 4);
  int*      redg  = (int*)     alloc((size_t)EPL_ * 4);
  int*      ccnt  = (int*)     alloc(N_ * 4);
  int*      cptr  = (int*)     alloc((N_ + 1) * 4);
  int*      ccur  = (int*)     alloc(N_ * 4);
  int*      cedg  = (int*)     alloc((size_t)EPL_ * 4);
  float*    cnt2  = (float*)   alloc(K_ * 4);
  float*    agg2  = (float*)   alloc(KH * 4);
  float*    x3    = (float*)   alloc(KH * 4);
  float*    x4    = (float*)   alloc(KH * 4);
  float*    h     = (float*)   alloc((size_t)B_ * 1024 * 4);
  float*    hl    = (float*)   alloc((size_t)B_ * HID_ * 4);

  const int EPB = (EPL_ + 255) / 256;        // one-thread-per-edge grids
  const int GN  = (N_ * HID_ + 255) / 256;

  // ---- embed
  k_embed<<<N_, 256, 0, stream>>>(x_type, ndep, et, ed, x0);

  // ---- gconv0 + pool
  hipMemsetAsync(agg, 0, NH * 4, stream);
  hipMemsetAsync(cnt, 0, N_ * 4, stream);
  k_edge_agg<<<E_, 256, 0, stream>>>(src, dst, x0, agg, cnt);
  k_mean<<<GN, 256, 0, stream>>>(agg, cnt, N_ * HID_);
  k_gemm<<<(N_ + 31) / 32, 256, 0, stream>>>(agg, 256, 0, c0_Wr, 256, x0, 256, c0_Wo, 256,
                                             c0_br, x1, 256, N_, 256, 256, 1);
  k_pool<<<B_, 256, 0, stream>>>(x1, NG_, h, 0);

  // ---- gconv1 + pool
  hipMemsetAsync(agg, 0, NH * 4, stream);
  hipMemsetAsync(cnt, 0, N_ * 4, stream);
  k_edge_agg<<<E_, 256, 0, stream>>>(src, dst, x1, agg, cnt);
  k_mean<<<GN, 256, 0, stream>>>(agg, cnt, N_ * HID_);
  k_gemm<<<(N_ + 31) / 32, 256, 0, stream>>>(agg, 256, 0, c1_Wr, 256, x1, 256, c1_Wo, 256,
                                             c1_br, x2, 256, N_, 256, 256, 1);
  k_pool<<<B_, 256, 0, stream>>>(x2, NG_, h, 256);

  // ---- ASAP attention pooling on x2
  hipMemsetAsync(mxu, 0, NH * 4, stream);                  // key 0 < key(-FLT_MAX): safe init (loops present)
  k_mx<<<EPL_, 256, 0, stream>>>(src, dst, x2, mxu);
  k_unkey<<<GN, 256, 0, stream>>>(mxu, mxf, N_ * HID_);
  k_gemm<<<(N_ + 31) / 32, 256, 0, stream>>>(mxf, 256, 0, p_Wlin, 256, nullptr, 0, nullptr, 0,
                                             p_blin, q, 256, N_, 256, 256, 0);
  k_dot1<<<N_, 256, 0, stream>>>(q,  p_Watt,        an, p_batt);   // a[n] = q.Watt[0:256] + batt
  k_dot1<<<N_, 256, 0, stream>>>(x2, p_Watt + 256,  cn, nullptr);  // c[n] = x.Watt[256:512]
  hipMemsetAsync(smaxu, 0, N_ * 4, stream);
  k_smax<<<EPB, 256, 0, stream>>>(src, dst, an, cn, sE, smaxu);
  hipMemsetAsync(denom, 0, N_ * 4, stream);
  k_exp<<<EPB, 256, 0, stream>>>(src, dst, sE, smaxu, eE, denom);
  k_p<<<EPB, 256, 0, stream>>>(dst, eE, denom, pE);
  hipMemsetAsync(xnew, 0, NH * 4, stream);
  k_xnew<<<EPL_, 256, 0, stream>>>(src, dst, x2, pE, xnew);
  k_dot1<<<N_, 256, 0, stream>>>(xnew, le_W2, av,  nullptr);
  k_dot1<<<N_, 256, 0, stream>>>(xnew, le_W3, bm,  le_b3);
  k_dot1<<<N_, 256, 0, stream>>>(xnew, le_W1, w1d, nullptr);
  hipMemsetAsync(fita, 0, N_ * 4, stream);
  k_fitacc<<<EPB, 256, 0, stream>>>(src, dst, av, bm, fita);
  k_fitness<<<(N_ + 255) / 256, 256, 0, stream>>>(fita, w1d, le_b1, fitn);
  hipMemsetAsync(selc, 0, B_ * 4, stream);
  k_topk<<<B_, 256, 0, stream>>>(fitn, selc, perm, pos, fits);
  k_xsel<<<K_, 256, 0, stream>>>(xnew, perm, fits, xsel);

  // ---- A2 = Sp^T (A Sp), sparse path
  hipMemsetAsync(Ssel, 0, NK * 4, stream);
  k_Sscatter<<<EPB, 256, 0, stream>>>(src, dst, pos, pE, Ssel);
  hipMemsetAsync(rcnt, 0, N_ * 4, stream);
  k_hist<<<EPB, 256, 0, stream>>>(src, rcnt);
  k_scan<<<1, 256, 0, stream>>>(rcnt, rptr, rcur, N_);
  k_scat<<<EPB, 256, 0, stream>>>(src, rcur, redg);
  k_AM<<<N_, 256, 0, stream>>>(rptr, redg, dst, Ssel, Mb);
  hipMemsetAsync(ccnt, 0, N_ * 4, stream);
  k_hist<<<EPB, 256, 0, stream>>>(dst, ccnt);
  k_scan<<<1, 256, 0, stream>>>(ccnt, cptr, ccur, N_);
  k_scat<<<EPB, 256, 0, stream>>>(dst, ccur, cedg);
  k_A2<<<K_, 256, 0, stream>>>(perm, cptr, cedg, src, pE, Mb, A2);
  k_colcnt<<<(K_ + 255) / 256, 256, 0, stream>>>(A2, cnt2);

  // ---- gconv2 (dense) + pool
  k_gemm<<<(K_ + 31) / 32, 256, 0, stream>>>(A2, K_, 1, xsel, 256, nullptr, 0, nullptr, 0,
                                             nullptr, agg2, 256, K_, K_, 256, 0);  // agg2 = A2^T @ xsel
  k_mean<<<(K_ * HID_ + 255) / 256, 256, 0, stream>>>(agg2, cnt2, K_ * HID_);
  k_gemm<<<(K_ + 31) / 32, 256, 0, stream>>>(agg2, 256, 0, c2_Wr, 256, xsel, 256, c2_Wo, 256,
                                             c2_br, x3, 256, K_, 256, 256, 1);
  k_pool<<<B_, 256, 0, stream>>>(x3, KPER_, h, 512);

  // ---- gconv3 (dense) + pool
  k_gemm<<<(K_ + 31) / 32, 256, 0, stream>>>(A2, K_, 1, x3, 256, nullptr, 0, nullptr, 0,
                                             nullptr, agg2, 256, K_, K_, 256, 0);
  k_mean<<<(K_ * HID_ + 255) / 256, 256, 0, stream>>>(agg2, cnt2, K_ * HID_);
  k_gemm<<<(K_ + 31) / 32, 256, 0, stream>>>(agg2, 256, 0, c3_Wr, 256, x3, 256, c3_Wo, 256,
                                             c3_br, x4, 256, K_, 256, 256, 1);
  k_pool<<<B_, 256, 0, stream>>>(x4, KPER_, h, 768);

  // ---- head
  k_gemm<<<1, 256, 0, stream>>>(h, 1024, 0, lin1_W, 256, nullptr, 0, nullptr, 0,
                                lin1_b, hl, 256, 16, 1024, 256, 1);
  k_gemm<<<1, 256, 0, stream>>>(hl, 256, 0, pred_W, 128, nullptr, 0, nullptr, 0,
                                pred_b, out, 128, 16, 256, NCLS_, 0);
}
